// QuillanIntegratedModel_77464030151267
// MI455X (gfx1250) — matrix operand order repaired and checked by reading, not compiled, not run on hardware
//
#include <hip/hip_runtime.h>
#include <hip/hip_bf16.h>
#include <math.h>

// ---------------- model constants ----------------
#define LQ   512
#define DIM  768
#define DFF  3072
#define LAYERS 8
#define NEXP 8
#define TOPK 2
#define VOCAB 50257
#define KSZ  4
#define STEPS 2
#define RH   (DIM/2)

typedef __attribute__((ext_vector_type(16))) __bf16 v16bf;
typedef __attribute__((ext_vector_type(4)))  __bf16 v4bf;
typedef __attribute__((ext_vector_type(8)))  float  v8f;
typedef __attribute__((ext_vector_type(4)))  int    v4i;

#ifndef __has_builtin
#define __has_builtin(x) 0
#endif

// gfx1250 async global->LDS path (guarded: falls back to sync staging if absent)
#if __has_builtin(__builtin_amdgcn_global_load_async_to_lds_b128)
#define HAS_ASYNC_LDS 1
typedef __attribute__((address_space(1))) v4i* gv4i_p;   // pointer to int4 in global AS
typedef __attribute__((address_space(3))) v4i* lv4i_p;   // pointer to int4 in LDS AS
#else
#define HAS_ASYNC_LDS 0
#endif

#if HAS_ASYNC_LDS
#if __has_builtin(__builtin_amdgcn_s_wait_asynccnt)
#define WAIT_ASYNC0() __builtin_amdgcn_s_wait_asynccnt(0)
#else
#define WAIT_ASYNC0() asm volatile("s_wait_asynccnt 0x0" ::: "memory")
#endif
#else
#define WAIT_ASYNC0() ((void)0)
#endif

// ternary quantize with precomputed reciprocal scale: clip(round(w*invg), -1, 1)
__device__ __forceinline__ float ternq(float w, float invg) {
    return fminf(1.f, fmaxf(-1.f, rintf(w * invg)));
}

// ---------------- scale (absmean) reduction: deterministic two-stage ----------------
__global__ __launch_bounds__(256) void abssum_partial_k(const float* __restrict__ w, int n,
                                                        float* __restrict__ part) {
    __shared__ float red[256];
    float s = 0.f;
    for (int i = blockIdx.x * blockDim.x + threadIdx.x; i < n; i += gridDim.x * blockDim.x)
        s += fabsf(w[i]);
    red[threadIdx.x] = s; __syncthreads();
    for (int o = 128; o > 0; o >>= 1) {
        if (threadIdx.x < o) red[threadIdx.x] += red[threadIdx.x + o];
        __syncthreads();
    }
    if (threadIdx.x == 0) part[blockIdx.x] = red[0];
}

__global__ void abssum_final_k(const float* __restrict__ part, int nb, float inv_n,
                               float* __restrict__ out) {
    float s = 0.f;
    for (int i = 0; i < nb; ++i) s += part[i];
    out[0] = fmaxf(s * inv_n, 1e-5f);
}

// ---------------- bf16 WMMA GEMM: C = epilogue( scale * (A @ ternary(W).T) ) -----------
// A: [M x K] bf16 row-major.  W: [N x K] f32 row-major, ternary-quantized on the fly.
// block = 256 thr = 8 waves (2x4); block tile 64(M) x 128(N); K step 32.
// Double-buffered LDS, software-pipelined: stage tile k+1 (async-to-LDS for A when
// available) while 4 WMMAs consume tile k. One barrier per K-iteration.
#define EPI_F32      0   // Cf[..] = v
#define EPI_RELU_BF  1   // Cb[..] = bf16(max(v,0))
#define EPI_GELU_BF  2   // Cb[..] = bf16(gelu_exact(v))
#define EPI_RES_F32  3   // Cf[..] = v + res[..]
#define EPI_MASKACC  4   // if mask[row*ms] != 0 : Cf[..] += v

__global__ __launch_bounds__(256)
void gemm_wmma_bf16_k(const __bf16* __restrict__ A, const float* __restrict__ W,
                      const float* __restrict__ scale,
                      float* __restrict__ Cf, __bf16* __restrict__ Cb,
                      const float* __restrict__ res,
                      const float* __restrict__ mask, int mask_stride,
                      int M, int N, int K, int mode)
{
    __shared__ __align__(16) __bf16 lsA[2][64][32];    // 2 x 4 KB
    __shared__ __align__(16) __bf16 lsB[2][128][32];   // 2 x 8 KB

    const int tid  = threadIdx.x;
    const int wave = tid >> 5;
    const int lane = tid & 31;
    const int mt   = wave >> 2;          // 0..1
    const int nt   = wave & 3;           // 0..3
    const int kh   = lane >> 4;          // lane half
    const int ln15 = lane & 15;
    const int m0   = blockIdx.y * 64;
    const int n0   = blockIdx.x * 128;
    const float g    = scale[0];
    const float invg = 1.f / g;          // hoisted: single divide per thread

    const v8f zero8 = {0.f, 0.f, 0.f, 0.f, 0.f, 0.f, 0.f, 0.f};
    v8f acc[2][2];
    acc[0][0] = zero8; acc[0][1] = zero8; acc[1][0] = zero8; acc[1][1] = zero8;

    // A staging map: 64 rows x 64 B; 256 chunks of 16 B -> 1 per thread
    const int arow = tid >> 2;
    const int achk = (tid & 3) * 8;      // bf16 element offset

    // stage one 32-wide K tile into buffer p
    auto stage = [&](int sk, int p) {
        // ---- A tile (64x32 bf16): 16B per thread ----
        const __bf16* srcA = &A[(size_t)(m0 + arow) * K + sk + achk];
#if HAS_ASYNC_LDS
        __builtin_amdgcn_global_load_async_to_lds_b128(
            (gv4i_p)(void*)srcA,
            (lv4i_p)(void*)&lsA[p][arow][achk], 0, 0);
#else
        *(uint4*)&lsA[p][arow][achk] = *(const uint4*)srcA;
#endif
        // ---- B tile (128x32 f32 -> ternary bf16): float4 loads, 8B LDS stores ----
        #pragma unroll
        for (int j = 0; j < 4; ++j) {
            int c   = tid + 256 * j;     // 0..1023 float4-chunks
            int row = c >> 3;
            int q4  = (c & 7) * 4;
            int gn  = n0 + row;
            v4bf qv;
            if (gn < N) {
                const float4 w4 = *(const float4*)&W[(size_t)gn * K + sk + q4];
                qv[0] = (__bf16)ternq(w4.x, invg);
                qv[1] = (__bf16)ternq(w4.y, invg);
                qv[2] = (__bf16)ternq(w4.z, invg);
                qv[3] = (__bf16)ternq(w4.w, invg);
            } else {
                qv[0] = (__bf16)0.f; qv[1] = (__bf16)0.f;
                qv[2] = (__bf16)0.f; qv[3] = (__bf16)0.f;
            }
            *(v4bf*)&lsB[p][row][q4] = qv;
        }
        // prefetch the tile after this one (gfx1250 global_prefetch_b8)
        if (sk + 32 < K) {
            int pn = n0 + (tid >> 1);
            if (pn < N) __builtin_prefetch(&W[(size_t)pn * K + sk + 32], 0, 1);
        }
    };

    // prologue: stage tile 0
    stage(0, 0);
    WAIT_ASYNC0();
    __syncthreads();

    #pragma unroll 2
    for (int k0 = 0; k0 < K; k0 += 32) {
        const int p = (k0 >> 5) & 1;
        // stage next tile into the other buffer while computing this one
        if (k0 + 32 < K) stage(k0 + 32, p ^ 1);

        // ---- fragments (documented CDNA5 16-bit A 16x32 / B 32x16 layouts) ----
        v16bf af[2], bfr[2];
        #pragma unroll
        for (int mi = 0; mi < 2; ++mi) {
            const int mrow = mt * 32 + mi * 16 + ln15;
            #pragma unroll
            for (int v = 0; v < 4; ++v) {
                af[mi][2*v]     = lsA[p][mrow][kh*8 + 2*v];
                af[mi][2*v + 1] = lsA[p][mrow][kh*8 + 2*v + 1];
                af[mi][2*v + 8] = lsA[p][mrow][16 + kh*8 + 2*v];
                af[mi][2*v + 9] = lsA[p][mrow][16 + kh*8 + 2*v + 1];
            }
        }
        #pragma unroll
        for (int ni = 0; ni < 2; ++ni) {
            const int nrow = nt * 32 + ni * 16 + ln15;
            #pragma unroll
            for (int v = 0; v < 8; ++v) {
                bfr[ni][2*v]     = lsB[p][nrow][kh*8 + 2*v + (v >= 4 ? 8 : 0)];
                bfr[ni][2*v + 1] = lsB[p][nrow][kh*8 + 2*v + 1 + (v >= 4 ? 8 : 0)];
            }
        }

        #pragma unroll
        for (int mi = 0; mi < 2; ++mi)
            #pragma unroll
            for (int ni = 0; ni < 2; ++ni)
                acc[mi][ni] = __builtin_amdgcn_wmma_f32_16x16x32_bf16(
                    false, af[mi], false, bfr[ni], (short)0, acc[mi][ni], false, false);

        // next tile staged (incl. async A) + everyone done reading buffer p
        WAIT_ASYNC0();
        __syncthreads();
    }

    // ---- epilogue: lane holds rows m = 8*kh + v, col = lane&15 of each 16x16 tile ----
    #pragma unroll
    for (int mi = 0; mi < 2; ++mi) {
        const int rbase = m0 + mt * 32 + mi * 16 + 8 * kh;
        #pragma unroll
        for (int ni = 0; ni < 2; ++ni) {
            const int col = n0 + nt * 32 + ni * 16 + ln15;
            if (col >= N) continue;
            #pragma unroll
            for (int v = 0; v < 8; ++v) {
                const int row = rbase + v;
                float val = acc[mi][ni][v] * g;
                size_t off2 = (size_t)row * N + col;
                switch (mode) {
                    case EPI_F32:     Cf[off2] = val; break;
                    case EPI_RELU_BF: Cb[off2] = (__bf16)fmaxf(val, 0.f); break;
                    case EPI_GELU_BF: Cb[off2] = (__bf16)(0.5f * val *
                                          (1.f + erff(val * 0.70710678118f))); break;
                    case EPI_RES_F32: Cf[off2] = val + res[off2]; break;
                    case EPI_MASKACC: if (mask[(size_t)row * mask_stride] != 0.f)
                                          Cf[off2] += val; break;
                }
            }
        }
    }
}

// ---------------- element-wise / small kernels ----------------
__global__ void embed_gather_k(const int* __restrict__ ids, const float* __restrict__ emb,
                               float* __restrict__ x) {
    int idx = blockIdx.x * blockDim.x + threadIdx.x;
    if (idx >= LQ * DIM) return;
    int l = idx / DIM, d = idx % DIM;
    x[idx] = emb[(size_t)ids[l] * DIM + d];
}

__global__ __launch_bounds__(256) void rmsnorm_bf16_k(const float* __restrict__ x,
                                                      const float* __restrict__ w,
                                                      __bf16* __restrict__ out) {
    int l = blockIdx.x;
    __shared__ float red[256];
    float s = 0.f;
    for (int d = threadIdx.x; d < DIM; d += 256) { float v = x[l*DIM + d]; s += v * v; }
    red[threadIdx.x] = s; __syncthreads();
    for (int o = 128; o > 0; o >>= 1) {
        if (threadIdx.x < o) red[threadIdx.x] += red[threadIdx.x + o];
        __syncthreads();
    }
    float r = rsqrtf(red[0] / (float)DIM + 1e-6f);
    for (int d = threadIdx.x; d < DIM; d += 256)
        out[l*DIM + d] = (__bf16)(w[d] * x[l*DIM + d] * r);
}

__global__ void conv_gate_k(const float* __restrict__ proj, const float* __restrict__ cw,
                            const float* __restrict__ cb, __bf16* __restrict__ mix) {
    int idx = blockIdx.x * blockDim.x + threadIdx.x;
    if (idx >= LQ * DIM) return;
    int l = idx / DIM, d = idx % DIM;
    float acc = cb[d];
    #pragma unroll
    for (int k = 0; k < KSZ; ++k) {
        int ls = l - (KSZ - 1) + k;
        if (ls >= 0) acc += cw[d * KSZ + k] * proj[(size_t)ls * (2*DIM) + d];
    }
    float gin = proj[(size_t)l * (2*DIM) + DIM + d];
    float gate = gin / (1.f + expf(-gin));           // silu
    mix[idx] = (__bf16)(acc * gate);
}

__global__ void f32_to_bf16_k(const float* __restrict__ in, __bf16* __restrict__ out, int n) {
    int i = blockIdx.x * blockDim.x + threadIdx.x;
    if (i < n) out[i] = (__bf16)in[i];
}

__global__ void zero_f32_k(float* __restrict__ p, int n) {
    int i = blockIdx.x * blockDim.x + threadIdx.x;
    if (i < n) p[i] = 0.f;
}

__global__ void complexity_k(const __bf16* __restrict__ rh, const float* __restrict__ rw2,
                             const float* __restrict__ scales, float* __restrict__ comp) {
    int l = blockIdx.x * blockDim.x + threadIdx.x;
    if (l >= LQ) return;
    float g = scales[17];
    float invg = 1.f / g;
    float a = 0.f;
    for (int j = 0; j < RH; ++j) a += (float)rh[(size_t)l * RH + j] * ternq(rw2[j], invg);
    a *= g;
    comp[l] = 1.f / (1.f + expf(-a));
}

__global__ __launch_bounds__(256) void meanflag_k(const float* __restrict__ comp,
                                                  float* __restrict__ flag) {
    __shared__ float red[256];
    float s = 0.f;
    for (int i = threadIdx.x; i < LQ; i += 256) s += comp[i];
    red[threadIdx.x] = s; __syncthreads();
    for (int o = 128; o > 0; o >>= 1) {
        if (threadIdx.x < o) red[threadIdx.x] += red[threadIdx.x + o];
        __syncthreads();
    }
    if (threadIdx.x == 0) flag[0] = (red[0] / (float)LQ > 0.1f) ? 1.f : 0.f;
}

__global__ void ctx_prep_k(const float* __restrict__ cur, const float* __restrict__ temb,
                           float* __restrict__ ctxf, __bf16* __restrict__ ctxb) {
    int idx = blockIdx.x * blockDim.x + threadIdx.x;
    if (idx >= LQ * DIM) return;
    float v = cur[idx] + temb[idx % DIM];
    ctxf[idx] = v;
    ctxb[idx] = (__bf16)v;
}

__global__ void gate_topk_k(const float* __restrict__ ctx, const float* __restrict__ gw,
                            const float* __restrict__ scales, float* __restrict__ mask) {
    int l = blockIdx.x * blockDim.x + threadIdx.x;
    if (l >= LQ) return;
    float g = scales[18];
    float invg = 1.f / g;
    float logit[NEXP];
    for (int e = 0; e < NEXP; ++e) {
        float a = 0.f;
        for (int d = 0; d < DIM; ++d)
            a += ctx[(size_t)l * DIM + d] * ternq(gw[(size_t)e * DIM + d], invg);
        logit[e] = a * g;
    }
    int b0 = 0;
    for (int e = 1; e < NEXP; ++e) if (logit[e] > logit[b0]) b0 = e;
    int b1 = -1;
    for (int e = 0; e < NEXP; ++e) {
        if (e == b0) continue;
        if (b1 < 0 || logit[e] > logit[b1]) b1 = e;
    }
    for (int e = 0; e < NEXP; ++e)
        mask[(size_t)l * NEXP + e] = (e == b0 || e == b1) ? 1.f : 0.f;
}

__global__ __launch_bounds__(256) void step_update_k(const float* __restrict__ flat,
                                                     float* __restrict__ cur,
                                                     const float* __restrict__ cnw,
                                                     const float* __restrict__ comp) {
    int l = blockIdx.x;
    __shared__ float red[256];
    float s = 0.f;
    for (int d = threadIdx.x; d < DIM; d += 256) {
        float v = flat[l*DIM + d] + cur[l*DIM + d];
        s += v * v;
    }
    red[threadIdx.x] = s; __syncthreads();
    for (int o = 128; o > 0; o >>= 1) {
        if (threadIdx.x < o) red[threadIdx.x] += red[threadIdx.x + o];
        __syncthreads();
    }
    float r = rsqrtf(red[0] / (float)DIM + 1e-6f);
    float hard = (comp[l] > 0.5f) ? 1.f : 0.f;
    for (int d = threadIdx.x; d < DIM; d += 256) {
        float v = flat[l*DIM + d] + cur[l*DIM + d];
        float refined = cnw[d] * v * r;
        cur[l*DIM + d] = hard * refined + (1.f - hard) * cur[l*DIM + d];
    }
}

__global__ void final_sel_k(const float* __restrict__ cur, const float* __restrict__ core,
                            const float* __restrict__ flag, __bf16* __restrict__ out) {
    int i = blockIdx.x * blockDim.x + threadIdx.x;
    if (i >= LQ * DIM) return;
    out[i] = (__bf16)((flag[0] > 0.5f) ? cur[i] : core[i]);
}

__global__ void copy_comp_k(const float* __restrict__ comp, float* __restrict__ dst) {
    int l = blockIdx.x * blockDim.x + threadIdx.x;
    if (l < LQ) dst[l] = comp[l];
}

// ---------------- host orchestration ----------------
static inline size_t alignup(size_t x) { return (x + 255) & ~(size_t)255; }

extern "C" void kernel_launch(void* const* d_in, const int* in_sizes, int n_in,
                              void* d_out, int out_size, void* d_ws, size_t ws_size,
                              hipStream_t stream) {
    (void)in_sizes; (void)n_in; (void)out_size; (void)ws_size;

    const int*   ids   = (const int*)  d_in[0];
    const float* embed = (const float*)d_in[1];
    const float* normw = (const float*)d_in[2];
    const float* inpw  = (const float*)d_in[3];
    const float* convw = (const float*)d_in[4];
    const float* convb = (const float*)d_in[5];
    const float* outpw = (const float*)d_in[6];
    const float* rw1   = (const float*)d_in[7];
    const float* rw2   = (const float*)d_in[8];
    const float* gatew = (const float*)d_in[9];
    const float* temb  = (const float*)d_in[10];
    const float* ew1   = (const float*)d_in[11];
    const float* ew2   = (const float*)d_in[12];
    const float* cnw   = (const float*)d_in[13];
    const float* headw = (const float*)d_in[14];
    float* out = (float*)d_out;

    // workspace carve
    char* ws = (char*)d_ws;
    size_t off = 0;
    float* scales   = (float*)(ws + off); off = alignup(off + 64 * 4);        // [0..35]=g, [40]=flag
    float* partials = (float*)(ws + off); off = alignup(off + 256 * 4);
    float* X        = (float*)(ws + off); off = alignup(off + (size_t)LQ*DIM*4);
    float* CUR      = (float*)(ws + off); off = alignup(off + (size_t)LQ*DIM*4);
    float* PROJ     = (float*)(ws + off); off = alignup(off + (size_t)LQ*2*DIM*4);
    float* CTXF     = (float*)(ws + off); off = alignup(off + (size_t)LQ*DIM*4);
    float* FLATOUT  = (float*)(ws + off); off = alignup(off + (size_t)LQ*DIM*4);
    float* COMP     = (float*)(ws + off); off = alignup(off + (size_t)LQ*4);
    float* MASK     = (float*)(ws + off); off = alignup(off + (size_t)LQ*NEXP*4);
    __bf16* HBF     = (__bf16*)(ws + off); off = alignup(off + (size_t)LQ*DIM*2);
    __bf16* MIXBF   = (__bf16*)(ws + off); off = alignup(off + (size_t)LQ*DIM*2);
    __bf16* RHBF    = (__bf16*)(ws + off); off = alignup(off + (size_t)LQ*RH*2);
    __bf16* CTXBF   = (__bf16*)(ws + off); off = alignup(off + (size_t)LQ*DIM*2);
    __bf16* HMIDBF  = (__bf16*)(ws + off); off = alignup(off + (size_t)LQ*DFF*2);
    __bf16* COREBF  = (__bf16*)(ws + off); off = alignup(off + (size_t)LQ*DIM*2);
    __bf16* HEADBF  = (__bf16*)(ws + off); off = alignup(off + (size_t)LQ*DIM*2);

    const int ETHR = 256;
    const int NLD  = LQ * DIM;               // 393216
    const int ldb  = (NLD + ETHR - 1) / ETHR;

    // ---- per-tensor absmean scales (deterministic two-stage) ----
    auto scale_for = [&](const float* w, size_t n, int slot) {
        abssum_partial_k<<<256, 256, 0, stream>>>(w, (int)n, partials);
        abssum_final_k<<<1, 1, 0, stream>>>(partials, 256, 1.f / (float)n, scales + slot);
    };
    for (int i = 0; i < LAYERS; ++i) scale_for(inpw  + (size_t)i*2*DIM*DIM, (size_t)2*DIM*DIM, i);
    for (int i = 0; i < LAYERS; ++i) scale_for(outpw + (size_t)i*DIM*DIM,   (size_t)DIM*DIM, 8 + i);
    scale_for(rw1, (size_t)RH*DIM, 16);
    scale_for(rw2, (size_t)RH, 17);
    scale_for(gatew, (size_t)NEXP*DIM, 18);
    for (int e = 0; e < NEXP; ++e) scale_for(ew1 + (size_t)e*DFF*DIM, (size_t)DFF*DIM, 19 + e);
    for (int e = 0; e < NEXP; ++e) scale_for(ew2 + (size_t)e*DIM*DFF, (size_t)DIM*DFF, 27 + e);
    scale_for(headw, (size_t)VOCAB*DIM, 35);

    auto gemm = [&](const __bf16* A, const float* W, int slot,
                    float* Cf, __bf16* Cb, const float* res,
                    const float* mask, int ms, int M, int N, int K, int mode) {
        dim3 grid((N + 127) / 128, M / 64);
        gemm_wmma_bf16_k<<<grid, 256, 0, stream>>>(A, W, scales + slot, Cf, Cb, res,
                                                   mask, ms, M, N, K, mode);
    };

    // ---- embedding ----
    embed_gather_k<<<ldb, ETHR, 0, stream>>>(ids, embed, X);

    // ---- Mamba-style BitNet layers ----
    for (int i = 0; i < LAYERS; ++i) {
        rmsnorm_bf16_k<<<LQ, 256, 0, stream>>>(X, normw + (size_t)i*DIM, HBF);
        gemm(HBF, inpw + (size_t)i*2*DIM*DIM, i, PROJ, nullptr, nullptr, nullptr, 0,
             LQ, 2*DIM, DIM, EPI_F32);
        conv_gate_k<<<ldb, ETHR, 0, stream>>>(PROJ, convw + (size_t)i*DIM*KSZ,
                                              convb + (size_t)i*DIM, MIXBF);
        gemm(MIXBF, outpw + (size_t)i*DIM*DIM, 8 + i, X, nullptr, X, nullptr, 0,
             LQ, DIM, DIM, EPI_RES_F32);
    }
    // X is now "core"
    (void)hipMemcpyAsync(CUR, X, (size_t)LQ*DIM*4, hipMemcpyDeviceToDevice, stream);
    f32_to_bf16_k<<<ldb, ETHR, 0, stream>>>(X, COREBF, NLD);

    // ---- ComplexityRouter ----
    gemm(COREBF, rw1, 16, nullptr, RHBF, nullptr, nullptr, 0, LQ, RH, DIM, EPI_RELU_BF);
    complexity_k<<<(LQ + 255) / 256, 256, 0, stream>>>(RHBF, rw2, scales, COMP);
    meanflag_k<<<1, 256, 0, stream>>>(COMP, scales + 40);

    // ---- DiffusionCouncil ----
    for (int t = 0; t < STEPS; ++t) {
        ctx_prep_k<<<ldb, ETHR, 0, stream>>>(CUR, temb + (size_t)t*DIM, CTXF, CTXBF);
        gate_topk_k<<<(LQ + 255) / 256, 256, 0, stream>>>(CTXF, gatew, scales, MASK);
        zero_f32_k<<<ldb, ETHR, 0, stream>>>(FLATOUT, NLD);
        for (int e = 0; e < NEXP; ++e) {
            gemm(CTXBF, ew1 + (size_t)e*DFF*DIM, 19 + e, nullptr, HMIDBF, nullptr,
                 nullptr, 0, LQ, DFF, DIM, EPI_GELU_BF);
            gemm(HMIDBF, ew2 + (size_t)e*DIM*DFF, 27 + e, FLATOUT, nullptr, nullptr,
                 MASK + e, NEXP, LQ, DIM, DFF, EPI_MASKACC);
        }
        step_update_k<<<LQ, 256, 0, stream>>>(FLATOUT, CUR, cnw, COMP);
    }

    // ---- final select + LM head ----
    final_sel_k<<<ldb, ETHR, 0, stream>>>(CUR, X, scales + 40, HEADBF);
    gemm(HEADBF, headw, 35, out, nullptr, nullptr, nullptr, 0, LQ, VOCAB, DIM, EPI_F32);
    copy_comp_k<<<(LQ + 255) / 256, 256, 0, stream>>>(COMP, out + (size_t)LQ * VOCAB);
}
